// DeformableCrossAttentionImgWithPoint_81973745811804
// MI455X (gfx1250) — matrix operand order repaired and checked by reading, not compiled, not run on hardware
//
#include <hip/hip_runtime.h>
#include <hip/hip_bf16.h>
#include <cstdint>
#include <cstddef>

typedef __attribute__((ext_vector_type(16))) _Float16 v16h;
typedef __attribute__((ext_vector_type(8)))  _Float16 v8h;
typedef __attribute__((ext_vector_type(8)))  float    v8f;

namespace {
constexpr int kB  = 2;
constexpr int kNQ = 8192;
constexpr int kDM = 256;
constexpr int kNH = 6;
constexpr int kDH = 64;
constexpr int kNL = 3;
constexpr int kNP = 9;
constexpr int kBQ = kB * kNQ;                   // 16384
constexpr int kHW0 = 48 * 160;                  // 7680
constexpr int kHW1 = 24 * 80;                   // 1920
constexpr int kHW2 = 12 * 40;                   // 480
constexpr int kTOT = kHW0 + kHW1 + kHW2;        // 10080
constexpr int kNVal = kNH * kDH;                // 384
constexpr int kNOff = kNH * kNL * kNP * 2;      // 324
constexpr int kNAtt = kNH * kNL * kNP;          // 162
constexpr int kPts  = kNL * kNP;                // 27
}

// ---------------------------------------------------------------------------
// Elementwise prep kernels
// ---------------------------------------------------------------------------

// q = f16(x + pe), row-major [BQ][DM]
__global__ void add_to_f16(const float* __restrict__ x, const float* __restrict__ pe,
                           _Float16* __restrict__ q, int n) {
  int i = blockIdx.x * blockDim.x + threadIdx.x;
  if (i < n) q[i] = (_Float16)(x[i] + pe[i]);
}

// W[K][N] f32 (row-major) -> Wt[N][K] f16 (row-major, K contiguous)
__global__ void transpose_to_f16(const float* __restrict__ W, _Float16* __restrict__ Wt,
                                 int K, int N) {
  int i = blockIdx.x * blockDim.x + threadIdx.x;
  if (i >= K * N) return;
  int n = i / K, k = i % K;
  Wt[(size_t)n * K + k] = (_Float16)W[(size_t)k * N + n];
}

// feats (B, DM, h, w) channel-major -> src_f16[b][pix][DM] pixel-major (levels concatenated)
__global__ void feat_to_f16(const float* __restrict__ f0, const float* __restrict__ f1,
                            const float* __restrict__ f2, _Float16* __restrict__ src) {
  int i = blockIdx.x * blockDim.x + threadIdx.x;
  if (i >= kB * kTOT * kDM) return;
  int c = i % kDM;
  int rest = i / kDM;
  int pix = rest % kTOT;
  int b = rest / kTOT;
  const float* fp; int pl, hw;
  if (pix < kHW0)              { fp = f0; pl = pix;                hw = kHW0; }
  else if (pix < kHW0 + kHW1)  { fp = f1; pl = pix - kHW0;         hw = kHW1; }
  else                         { fp = f2; pl = pix - kHW0 - kHW1;  hw = kHW2; }
  src[i] = (_Float16)fp[((size_t)b * kDM + c) * hw + pl];
}

// per-query projected reference points, scaled per level: ref[bq][l][2]
__global__ void ref_points(const float* __restrict__ coor, const float* __restrict__ cam2img,
                           const float* __restrict__ lidar2cam, float* __restrict__ ref) {
  int t = blockIdx.x * blockDim.x + threadIdx.x;
  if (t >= kBQ) return;
  int b = t / kNQ;
  const float* pt = coor + (size_t)t * 3;
  const float* L  = lidar2cam + (size_t)b * 16;
  const float* Kc = cam2img + (size_t)b * 16;
  float x = pt[0], y = pt[1], z = pt[2];
  float cx = L[0]*x + L[1]*y + L[2]*z  + L[3];
  float cy = L[4]*x + L[5]*y + L[6]*z  + L[7];
  float cz = L[8]*x + L[9]*y + L[10]*z + L[11];
  float px = Kc[0]*cx + Kc[1]*cy + Kc[2]*cz  + Kc[3];
  float py = Kc[4]*cx + Kc[5]*cy + Kc[6]*cz  + Kc[7];
  float pz = Kc[8]*cx + Kc[9]*cy + Kc[10]*cz + Kc[11];
  float invz = 1.0f / pz;
  float u = px * invz, v = py * invz;
  const float sw[3] = {160.0f/1280.0f, 80.0f/1280.0f, 40.0f/1280.0f};
  const float sh[3] = {48.0f/384.0f,   24.0f/384.0f,  12.0f/384.0f};
  float* rp = ref + (size_t)t * (kNL * 2);
  #pragma unroll
  for (int l = 0; l < kNL; ++l) { rp[l*2+0] = u * sw[l]; rp[l*2+1] = v * sh[l]; }
}

// ---------------------------------------------------------------------------
// WMMA GEMM: C[M][N] = A[M][K](f16,row-major) * Bt[N][K]^T(f16) + bias[N]; f32 out
// One 32x64 macro-tile per wave32 (2 M-subtiles x 4 N-subtiles), K-step 32:
// per step 12x b128 loads feed 8x v_wmma_f32_16x16x32_f16.
// Out-of-range rows/cols are CLAMPED for loads (always in-bounds, no exec
// divergence in the hot loop); garbage lanes are masked at the store.
// ---------------------------------------------------------------------------
__global__ void gemm_wmma_f16(const _Float16* __restrict__ A,
                              const _Float16* __restrict__ Bt,
                              const float* __restrict__ bias,
                              float* __restrict__ C,
                              int M, int N, int K) {
  const int lane = threadIdx.x & 31;
  const int wave = threadIdx.x >> 5;
  const int wavesPerBlock = blockDim.x >> 5;
  const int groupsN = (N + 63) >> 6;   // 64-wide column groups
  const int groupsM = (M + 31) >> 5;   // 32-tall row groups
  int g = blockIdx.x * wavesPerBlock + wave;
  if (g >= groupsM * groupsN) return;  // wave-uniform exit
  int gm = g / groupsN, gn = g % groupsN;
  int lm = lane & 15;
  // A frag (16x32 f16): lanes 0-15 own K {0..7,16..23}; lanes 16-31 own {8..15,24..31}
  int ka = (lane < 16) ? 0 : 8;
  // B frag (32x16 f16): lanes 0-15 own K {0..7,16..23} of col N=lane%16; lanes 16-31 own {8..15,24..31}
  int kb = (lane < 16) ? 0 : 8;

  int row0 = gm * 32 + lm;
  int row1 = row0 + 16;
  int r0c = row0 < M ? row0 : M - 1;
  int r1c = row1 < M ? row1 : M - 1;
  const _Float16* ap0 = A + (size_t)r0c * K + ka;
  const _Float16* ap1 = A + (size_t)r1c * K + ka;

  const _Float16* bp[4];
  int ncol[4];
  #pragma unroll
  for (int t = 0; t < 4; ++t) {
    int c = gn * 64 + t * 16 + lm;
    ncol[t] = c;
    int cc = c < N ? c : N - 1;
    bp[t] = Bt + (size_t)cc * K + kb;
  }

  v8f acc0[4] = {{}, {}, {}, {}};
  v8f acc1[4] = {{}, {}, {}, {}};
  for (int k0 = 0; k0 < K; k0 += 32) {
    union { v16h v; v8h h[2]; } a0, a1;
    a0.h[0] = *(const v8h*)(ap0 + k0);        // K = k0+ka .. +7
    a0.h[1] = *(const v8h*)(ap0 + k0 + 16);   // K = k0+ka+16 .. +23
    a1.h[0] = *(const v8h*)(ap1 + k0);
    a1.h[1] = *(const v8h*)(ap1 + k0 + 16);
    #pragma unroll
    for (int t = 0; t < 4; ++t) {
      union { v16h v; v8h h[2]; } bf;
      bf.h[0] = *(const v8h*)(bp[t] + k0);    // K = k0+kb .. +7
      bf.h[1] = *(const v8h*)(bp[t] + k0 + 16);// K = k0+kb+16 .. +23
      acc0[t] = __builtin_amdgcn_wmma_f32_16x16x32_f16(false, a0.v, false, bf.v,
                                                       (short)0, acc0[t], false, false);
      acc1[t] = __builtin_amdgcn_wmma_f32_16x16x32_f16(false, a1.v, false, bf.v,
                                                       (short)0, acc1[t], false, false);
    }
  }

  // C/D layout: vgpr j -> row base+(lane<16?0:8)+j, col = subtile base + lane%16
  int mb = gm * 32 + ((lane < 16) ? 0 : 8);
  #pragma unroll
  for (int t = 0; t < 4; ++t) {
    int n = ncol[t];
    if (n >= N) continue;
    float bvv = bias ? bias[n] : 0.0f;
    #pragma unroll
    for (int j = 0; j < 8; ++j) {
      int m0 = mb + j;
      if (m0 < M) C[(size_t)m0 * N + n] = acc0[t][j] + bvv;
      int m1 = mb + 16 + j;
      if (m1 < M) C[(size_t)m1 * N + n] = acc1[t][j] + bvv;
    }
  }
}

// ---------------------------------------------------------------------------
// In-place softmax over contiguous 27-element rows: attn[bq*NH + h][27]
// ---------------------------------------------------------------------------
__global__ void softmax27(float* __restrict__ attn, int rows) {
  int t = blockIdx.x * blockDim.x + threadIdx.x;
  if (t >= rows) return;
  float* p = attn + (size_t)t * kPts;
  float mx = -1e30f;
  #pragma unroll
  for (int i = 0; i < kPts; ++i) mx = fmaxf(mx, p[i]);
  float e[kPts], s = 0.0f;
  #pragma unroll
  for (int i = 0; i < kPts; ++i) { e[i] = __expf(p[i] - mx); s += e[i]; }
  float inv = 1.0f / s;
  #pragma unroll
  for (int i = 0; i < kPts; ++i) p[i] = e[i] * inv;
}

// ---------------------------------------------------------------------------
// Bilinear gather + attention-weighted accumulate.
// One 64-thread block per (bq, head); lane d owns channel d. Taps are
// coalesced 256B reads from v (31MB -> resident in the 192MB L2).
// ---------------------------------------------------------------------------
__global__ void sample_accum(const float* __restrict__ ref,      // [BQ][NL][2]
                             const float* __restrict__ off,      // [BQ][NH][NL][NP][2]
                             const float* __restrict__ attn,     // [BQ][NH][27]
                             const float* __restrict__ v,        // [B][TOT][NVal]
                             _Float16* __restrict__ midf16) {    // [BQ][NVal]
  int blk = blockIdx.x;
  int h  = blk % kNH;
  int bq = blk / kNH;
  int b  = bq / kNQ;
  int d  = threadIdx.x;  // 0..63
  const int Hs[3]  = {48, 24, 12};
  const int Ws[3]  = {160, 80, 40};
  const int Po[3]  = {0, kHW0, kHW0 + kHW1};
  const float* rp  = ref + (size_t)bq * (kNL * 2);
  const float* op  = off + (size_t)bq * kNOff + (size_t)h * (kNL * kNP * 2);
  const float* apr = attn + ((size_t)bq * kNH + h) * kPts;
  const float* vb  = v + (size_t)b * kTOT * kNVal + (size_t)h * kDH + d;
  float acc = 0.0f;
  #pragma unroll
  for (int l = 0; l < kNL; ++l) {
    const int wl = Ws[l], hl = Hs[l];
    const float rx = rp[l*2+0], ry = rp[l*2+1];
    const float* vl = vb + (size_t)Po[l] * kNVal;
    #pragma unroll
    for (int p = 0; p < kNP; ++p) {
      float a = apr[l * kNP + p];
      // loc = (ref+off)/[w,h]; x = loc_x*w - 0.5 == rx + off_x - 0.5
      float x = rx + op[(l*kNP + p)*2 + 0] - 0.5f;
      float y = ry + op[(l*kNP + p)*2 + 1] - 0.5f;
      float xf = floorf(x), yf = floorf(y);
      int x0 = (int)xf, y0 = (int)yf;
      float fx = x - xf, fy = y - yf;
      float s = 0.0f;
      {
        int xi = x0, yi = y0;
        if (xi >= 0 && xi < wl && yi >= 0 && yi < hl)
          s += (1.0f - fx) * (1.0f - fy) * vl[(size_t)(yi * wl + xi) * kNVal];
      }
      {
        int xi = x0 + 1, yi = y0;
        if (xi >= 0 && xi < wl && yi >= 0 && yi < hl)
          s += fx * (1.0f - fy) * vl[(size_t)(yi * wl + xi) * kNVal];
      }
      {
        int xi = x0, yi = y0 + 1;
        if (xi >= 0 && xi < wl && yi >= 0 && yi < hl)
          s += (1.0f - fx) * fy * vl[(size_t)(yi * wl + xi) * kNVal];
      }
      {
        int xi = x0 + 1, yi = y0 + 1;
        if (xi >= 0 && xi < wl && yi >= 0 && yi < hl)
          s += fx * fy * vl[(size_t)(yi * wl + xi) * kNVal];
      }
      acc += a * s;
    }
  }
  midf16[(size_t)bq * kNVal + (size_t)h * kDH + d] = (_Float16)acc;
}

// ---------------------------------------------------------------------------
// Host-side orchestration
// ---------------------------------------------------------------------------
extern "C" void kernel_launch(void* const* d_in, const int* in_sizes, int n_in,
                              void* d_out, int out_size, void* d_ws, size_t ws_size,
                              hipStream_t stream) {
  const float* x    = (const float*)d_in[0];
  const float* pe   = (const float*)d_in[1];
  const float* coor = (const float*)d_in[2];
  const float* c2i  = (const float*)d_in[3];
  const float* l2c  = (const float*)d_in[4];
  const float* f0   = (const float*)d_in[5];
  const float* f1   = (const float*)d_in[6];
  const float* f2   = (const float*)d_in[7];
  const float* Wv   = (const float*)d_in[8];
  const float* bv   = (const float*)d_in[9];
  const float* Wo   = (const float*)d_in[10];
  const float* bo   = (const float*)d_in[11];
  const float* Wa   = (const float*)d_in[12];
  const float* ba   = (const float*)d_in[13];
  const float* Wu   = (const float*)d_in[14];
  const float* bu   = (const float*)d_in[15];

  // workspace partition (~95 MB total), 256B-aligned regions
  uint8_t* ws = (uint8_t*)d_ws;
  size_t cur = 0;
  auto alloc = [&](size_t bytes) -> void* {
    void* p = ws + cur;
    cur = (cur + bytes + 255) & ~(size_t)255;
    return p;
  };
  _Float16* qf    = (_Float16*)alloc((size_t)kBQ * kDM * sizeof(_Float16));          // 8.4 MB
  _Float16* Wvt   = (_Float16*)alloc((size_t)kNVal * kDM * sizeof(_Float16));
  _Float16* Wot   = (_Float16*)alloc((size_t)kNOff * kDM * sizeof(_Float16));
  _Float16* Wat   = (_Float16*)alloc((size_t)kNAtt * kDM * sizeof(_Float16));
  _Float16* Wut   = (_Float16*)alloc((size_t)kDM * kNVal * sizeof(_Float16));
  _Float16* srcf  = (_Float16*)alloc((size_t)kB * kTOT * kDM * sizeof(_Float16));    // 10.3 MB
  float*    refb  = (float*)alloc((size_t)kBQ * kNL * 2 * sizeof(float));
  float*    offr  = (float*)alloc((size_t)kBQ * kNOff * sizeof(float));              // 21.2 MB
  float*    attnb = (float*)alloc((size_t)kBQ * kNAtt * sizeof(float));              // 10.6 MB
  float*    vbuf  = (float*)alloc((size_t)kB * kTOT * kNVal * sizeof(float));        // 31.0 MB
  _Float16* midf  = (_Float16*)alloc((size_t)kBQ * kNVal * sizeof(_Float16));        // 12.6 MB
  (void)ws_size; (void)in_sizes; (void)n_in; (void)out_size;

  const int TPB = 256;
  auto cdiv = [](int a, int b) { return (a + b - 1) / b; };

  // weights -> f16, N-major (B^T layout for WMMA)
  transpose_to_f16<<<cdiv(kDM * kNVal, TPB), TPB, 0, stream>>>(Wv, Wvt, kDM, kNVal);
  transpose_to_f16<<<cdiv(kDM * kNOff, TPB), TPB, 0, stream>>>(Wo, Wot, kDM, kNOff);
  transpose_to_f16<<<cdiv(kDM * kNAtt, TPB), TPB, 0, stream>>>(Wa, Wat, kDM, kNAtt);
  transpose_to_f16<<<cdiv(kNVal * kDM, TPB), TPB, 0, stream>>>(Wu, Wut, kNVal, kDM);

  // activations
  add_to_f16<<<cdiv(kBQ * kDM, TPB), TPB, 0, stream>>>(x, pe, qf, kBQ * kDM);
  feat_to_f16<<<cdiv(kB * kTOT * kDM, TPB), TPB, 0, stream>>>(f0, f1, f2, srcf);
  ref_points<<<cdiv(kBQ, TPB), TPB, 0, stream>>>(coor, c2i, l2c, refb);

  auto launch_gemm = [&](const _Float16* A, const _Float16* Bt, const float* bias,
                         float* C, int M, int N, int K) {
    int groups = cdiv(M, 32) * cdiv(N, 64);
    const int wavesPerBlock = 4;
    gemm_wmma_f16<<<cdiv(groups, wavesPerBlock), wavesPerBlock * 32, 0, stream>>>(
        A, Bt, bias, C, M, N, K);
  };

  // projections
  launch_gemm(qf, Wot, bo, offr, kBQ, kNOff, kDM);            // offsets
  launch_gemm(qf, Wat, ba, attnb, kBQ, kNAtt, kDM);           // attn logits
  softmax27<<<cdiv(kBQ * kNH, TPB), TPB, 0, stream>>>(attnb, kBQ * kNH);
  launch_gemm(srcf, Wvt, bv, vbuf, kB * kTOT, kNVal, kDM);    // value proj

  // gather + weighted accumulate -> f16 mid
  sample_accum<<<kBQ * kNH, kDH, 0, stream>>>(refb, offr, attnb, vbuf, midf);

  // output projection -> d_out (f32)
  launch_gemm(midf, Wut, bu, (float*)d_out, kBQ, kDM, kNVal);
}
